// MMRI_P2I_6828998000691
// MI455X (gfx1250) — matrix operand order repaired and checked by reading, not compiled, not run on hardware
//
#include <hip/hip_runtime.h>

typedef _Float16 half_t;
typedef _Float16 v16h __attribute__((ext_vector_type(16)));
typedef _Float16 v8h  __attribute__((ext_vector_type(8)));
typedef _Float16 v2h  __attribute__((ext_vector_type(2)));
typedef float    v8f  __attribute__((ext_vector_type(8)));
typedef unsigned int u32;
typedef u32 u32x4 __attribute__((ext_vector_type(4)));
typedef int i32x4 __attribute__((ext_vector_type(4)));
typedef int i32x8 __attribute__((ext_vector_type(8)));

#define NIMG 6
#define CCH  64
#define HH   112
#define WW   200
#define HWSZ (HH * WW)                 // 22400
#define TILES_PER_IMG (HWSZ / 16)      // 1400
#define NTILES (NIMG * TILES_PER_IMG)  // 8400

#if defined(__HIP_DEVICE_COMPILE__) && __has_builtin(__builtin_amdgcn_tensor_load_to_lds)
#define HAS_TDM 1
#else
#define HAS_TDM 0
#endif

// ---------------------------------------------------------------------------
// Fold eval-mode BN into the 1x1 convs:  y = relu(A x + c)
// ---------------------------------------------------------------------------
__global__ void fuse_weights_kernel(const float* __restrict__ w,
                                    const float* __restrict__ g,
                                    const float* __restrict__ b,
                                    const float* __restrict__ m,
                                    const float* __restrict__ v,
                                    half_t* __restrict__ wf,
                                    float* __restrict__ bias) {
  for (int i = threadIdx.x; i < 5 * 64 * 64; i += blockDim.x) {
    int l = i >> 12;
    int o = (i >> 6) & 63;
    float s = g[l * 64 + o] * rsqrtf(v[l * 64 + o] + 1e-5f);
    wf[i] = (half_t)(w[i] * s);
  }
  for (int i = threadIdx.x; i < 5 * 64; i += blockDim.x) {
    float s = g[i] * rsqrtf(v[i] + 1e-5f);
    bias[i] = b[i] - s * m[i];
  }
}

// ---------------------------------------------------------------------------
// Fused 1x1-conv chain via v_wmma_f32_16x16x32_f16. One wave per 16-pixel
// tile (grid-strided; A fragments live in VGPRs across tiles).
// TWO: second layer chained through a wave-local LDS transpose.
// ---------------------------------------------------------------------------
template <bool TWO>
__global__ __launch_bounds__(256)
void conv_wmma_kernel(const float* __restrict__ in,    // [NIMG,64,HW] f32
                      const half_t* __restrict__ wf,   // [5,64,64] f16 fused
                      const float* __restrict__ bias,  // [5,64] f32
                      v2h* __restrict__ out,           // [NIMG,32,HW] half2
                      int l0, int l1) {
  __shared__ half_t xpose[8][64 * 16];  // 2KB per wave
  const int lane = threadIdx.x & 31;
  const int wave = threadIdx.x >> 5;
  const int h  = lane >> 4;
  const int mn = lane & 15;

  // A fragments (16-bit A 16x32 layout): each lane's fragment is two
  // contiguous 8-half runs -> two b128 loads + shuffle.
  v16h a0[4][2], a1[4][2];
  for (int mt = 0; mt < 4; ++mt)
    for (int kt = 0; kt < 2; ++kt) {
      const int row = 16 * mt + mn;
      const half_t* wp0 = wf + (l0 * 64 + row) * 64 + 32 * kt + 8 * h;
      v8h lo = *(const v8h*)wp0;         // k = 32kt+8h   .. +7
      v8h hi = *(const v8h*)(wp0 + 16);  // k = 32kt+16+8h.. +7
      a0[mt][kt] = __builtin_shufflevector(lo, hi, 0, 1, 2, 3, 4, 5, 6, 7, 8,
                                           9, 10, 11, 12, 13, 14, 15);
      if (TWO) {
        const half_t* wp1 = wf + (l1 * 64 + row) * 64 + 32 * kt + 8 * h;
        v8h lo1 = *(const v8h*)wp1;
        v8h hi1 = *(const v8h*)(wp1 + 16);
        a1[mt][kt] = __builtin_shufflevector(lo1, hi1, 0, 1, 2, 3, 4, 5, 6, 7,
                                             8, 9, 10, 11, 12, 13, 14, 15);
      } else {
        a1[mt][kt] = v16h{};
      }
    }

  const int nw = (int)((gridDim.x * blockDim.x) >> 5);
  for (int t = (int)((blockIdx.x * blockDim.x + threadIdx.x) >> 5); t < NTILES;
       t += nw) {
    const int img = t / TILES_PER_IMG;
    const int p   = (t - img * TILES_PER_IMG) * 16 + mn;

    // prefetch next tile's activations (global_prefetch_b8)
    {
      int tn = t + nw;
      if (tn < NTILES) {
        int img2 = tn / TILES_PER_IMG;
        int p2 = (tn - img2 * TILES_PER_IMG) * 16 + mn;
        __builtin_prefetch(&in[(img2 * CCH + 16 * h) * HWSZ + p2], 0, 1);
      }
    }

    // B fragments (32x16): k = 32*kt + 8*h + (i&7) + 16*(i>>3), col n = mn (f32 -> f16)
    v16h bf[2];
    for (int kt = 0; kt < 2; ++kt) {
      v16h tb = {};
      for (int i = 0; i < 16; ++i) {
        int k = 32 * kt + 8 * h + (i & 7) + 16 * (i >> 3);
        tb[i] = (half_t)in[(img * CCH + k) * HWSZ + p];
      }
      bf[kt] = tb;
    }

    v8f acc[4];
    for (int mt = 0; mt < 4; ++mt) {
      v8f c = {};
      c = __builtin_amdgcn_wmma_f32_16x16x32_f16(false, a0[mt][0], false, bf[0],
                                                 (short)0, c, false, false);
      c = __builtin_amdgcn_wmma_f32_16x16x32_f16(false, a0[mt][1], false, bf[1],
                                                 (short)0, c, false, false);
      acc[mt] = c;
    }

    if (TWO) {
      // bias + relu, stash as half2 [c2][n] in wave-local LDS
      half_t* tb = &xpose[wave][0];
      for (int mt = 0; mt < 4; ++mt)
        for (int r = 0; r < 8; r += 2) {
          int o = 16 * mt + 8 * h + r;  // even
          float x0 = fmaxf(acc[mt][r]     + bias[l0 * 64 + o],     0.0f);
          float x1 = fmaxf(acc[mt][r + 1] + bias[l0 * 64 + o + 1], 0.0f);
          v2h pk = {(half_t)x0, (half_t)x1};
          *(v2h*)&tb[2 * ((o >> 1) * 16 + mn)] = pk;
        }
      __asm__ volatile("s_wait_dscnt 0" ::: "memory");  // wave-local xchg

      v16h b2[2];
      for (int kt = 0; kt < 2; ++kt) {
        v16h tt = {};
        for (int i = 0; i < 16; ++i) {
          int k = 32 * kt + 8 * h + (i & 7) + 16 * (i >> 3);
          tt[i] = tb[2 * ((k >> 1) * 16 + mn) + (k & 1)];
        }
        b2[kt] = tt;
      }
      for (int mt = 0; mt < 4; ++mt) {
        v8f c = {};
        c = __builtin_amdgcn_wmma_f32_16x16x32_f16(false, a1[mt][0], false,
                                                   b2[0], (short)0, c, false,
                                                   false);
        c = __builtin_amdgcn_wmma_f32_16x16x32_f16(false, a1[mt][1], false,
                                                   b2[1], (short)0, c, false,
                                                   false);
        for (int r = 0; r < 8; r += 2) {
          int o = 16 * mt + 8 * h + r;
          float x0 = fmaxf(c[r]     + bias[l1 * 64 + o],     0.0f);
          float x1 = fmaxf(c[r + 1] + bias[l1 * 64 + o + 1], 0.0f);
          v2h pk = {(half_t)x0, (half_t)x1};
          out[(img * 32 + (o >> 1)) * HWSZ + p] = pk;
        }
      }
    } else {
      for (int mt = 0; mt < 4; ++mt)
        for (int r = 0; r < 8; r += 2) {
          int o = 16 * mt + 8 * h + r;
          float x0 = fmaxf(acc[mt][r]     + bias[l0 * 64 + o],     0.0f);
          float x1 = fmaxf(acc[mt][r + 1] + bias[l0 * 64 + o + 1], 0.0f);
          v2h pk = {(half_t)x0, (half_t)x1};
          out[(img * 32 + (o >> 1)) * HWSZ + p] = pk;
        }
    }
  }
}

// ---------------------------------------------------------------------------
// TDM: load one 3-D tile (x=16 half2, y=16 rows, z=32 channel-pairs) from a
// packed [32][H][W] half2 buffer into LDS.  D# per ISA 08_async_tensor §8.
// ---------------------------------------------------------------------------
#if HAS_TDM
__device__ __forceinline__ void tdm_load_tile_16x16x32(const void* gptr,
                                                       u32 lds_byte_off) {
  unsigned long long ga = (unsigned long long)gptr;
  u32x4 g0;
  g0[0] = 1u;                               // count=1 (user mode)
  g0[1] = lds_byte_off;                     // lds_addr
  g0[2] = (u32)(ga & 0xffffffffu);          // global_addr[31:0]
  g0[3] = (u32)((ga >> 32) & 0x01ffffffu)   // global_addr[56:32]
        | (2u << 30);                       // type = 2 ("image")
  i32x8 g1;
  g1[0] = (2 << 16);                        // data_size = 4B; wg_mask = 0
  g1[1] = (int)(200u << 16);                // tensor_dim0[15:0] = W
  g1[2] = (int)(112u << 16);                // tensor_dim0 hi; tensor_dim1 = H
  g1[3] = (int)(16u << 16);                 // tensor_dim1 hi; tile_dim0 = 16
  g1[4] = (int)(16u | (32u << 16));         // tile_dim1 = 16; tile_dim2 = 32
  g1[5] = 200;                              // tensor_dim0_stride = W
  g1[6] = (int)(22400u << 16);              // dim0_str hi; dim1_stride = HW
  g1[7] = 0;                                // dim1_stride[47:16]
  i32x4 g2;
  g2[0] = 32;                               // tensor_dim2 = 32 channel-pairs
  g2[1] = 0; g2[2] = 0; g2[3] = 0;          // no dim3 / no iterate
  i32x4 g3 = {0, 0, 0, 0};
#if __clang_major__ >= 23
  i32x8 g4 = {0, 0, 0, 0, 0, 0, 0, 0};
  __builtin_amdgcn_tensor_load_to_lds(g0, g1, g2, g3, g4, 0);
#else
  __builtin_amdgcn_tensor_load_to_lds(g0, g1, g2, g3, 0);
#endif
}
#endif

// ---------------------------------------------------------------------------
// 9x9 local attention, 8x8 pixel tile / 64-thread block.
// ---------------------------------------------------------------------------
#define RAD 4
__global__ __launch_bounds__(64)
void local_attn_kernel(const v2h* __restrict__ q, const v2h* __restrict__ k,
                       const v2h* __restrict__ v, float* __restrict__ out) {
  __shared__ v2h  kv[32][16][16];  // [c2][py][px], 32 KB
  __shared__ float sims[81][64];   // ~20 KB
  const int tid = threadIdx.x;
  const int tx = tid & 7, ty = tid >> 3;
  const int img = blockIdx.z;
  const int gx = blockIdx.x * 8 + tx;
  const int gy = blockIdx.y * 8 + ty;
  const bool valid = (gx < WW);
  const int p = gy * WW + gx;

  int oy = 0, ox = 0;  // LDS index shift for clamped tile origin (TDM path)

#if HAS_TDM
  {
    int sy0 = blockIdx.y * 8 - RAD, sx0 = blockIdx.x * 8 - RAD;
    int syc = min(max(sy0, 0), HH - 16);
    int sxc = min(max(sx0, 0), WW - 16);
    oy = sy0 - syc;
    ox = sx0 - sxc;
    if (tid < 32) {  // wave 0 issues the tensor DMA
      const v2h* gp = k + ((size_t)(img * 32)) * HWSZ + syc * WW + sxc;
      tdm_load_tile_16x16x32((const void*)gp,
                             (u32)(unsigned long long)(uintptr_t)&kv[0][0][0]);
      __builtin_amdgcn_s_wait_tensorcnt(0);
    }
  }
#else
  for (int e = tid; e < 32 * 256; e += 64) {
    int c2 = e >> 8, pix = e & 255;
    int py = pix >> 4, px = pix & 15;
    int sy = blockIdx.y * 8 + py - RAD;
    int sx = blockIdx.x * 8 + px - RAD;
    v2h val = {};
    if (sy >= 0 && sy < HH && sx >= 0 && sx < WW)
      val = k[(img * 32 + c2) * HWSZ + sy * WW + sx];
    kv[c2][py][px] = val;
  }
#endif

  v2h q2[32];
  for (int c2 = 0; c2 < 32; ++c2) q2[c2] = v2h{};
  if (valid)
    for (int c2 = 0; c2 < 32; ++c2) q2[c2] = q[(img * 32 + c2) * HWSZ + p];
  __syncthreads();

  float mmax = -1e30f;
  if (valid) {
    int idx = 0;
    for (int di = 0; di < 9; ++di)
      for (int dj = 0; dj < 9; ++dj, ++idx) {
        float s = 0.0f;
        for (int c2 = 0; c2 < 32; ++c2) {
          v2h kk = kv[c2][ty + di + oy][tx + dj + ox];
          s += (float)q2[c2][0] * (float)kk[0] +
               (float)q2[c2][1] * (float)kk[1];
        }
        s *= 0.125f;  // 1/sqrt(64)
        sims[idx][tid] = s;
        mmax = fmaxf(mmax, s);
      }
  }
  __syncthreads();

  // restage VALUE halo into the same buffer
#if HAS_TDM
  {
    int sy0 = blockIdx.y * 8 - RAD, sx0 = blockIdx.x * 8 - RAD;
    int syc = min(max(sy0, 0), HH - 16);
    int sxc = min(max(sx0, 0), WW - 16);
    if (tid < 32) {
      const v2h* gp = v + ((size_t)(img * 32)) * HWSZ + syc * WW + sxc;
      tdm_load_tile_16x16x32((const void*)gp,
                             (u32)(unsigned long long)(uintptr_t)&kv[0][0][0]);
      __builtin_amdgcn_s_wait_tensorcnt(0);
    }
  }
#else
  for (int e = tid; e < 32 * 256; e += 64) {
    int c2 = e >> 8, pix = e & 255;
    int py = pix >> 4, px = pix & 15;
    int sy = blockIdx.y * 8 + py - RAD;
    int sx = blockIdx.x * 8 + px - RAD;
    v2h val = {};
    if (sy >= 0 && sy < HH && sx >= 0 && sx < WW)
      val = v[(img * 32 + c2) * HWSZ + sy * WW + sx];
    kv[c2][py][px] = val;
  }
#endif
  __syncthreads();

  if (valid) {
    float acc[64];
    for (int i = 0; i < 64; ++i) acc[i] = 0.0f;
    float sum = 0.0f;
    int idx = 0;
    for (int di = 0; di < 9; ++di)
      for (int dj = 0; dj < 9; ++dj, ++idx) {
        float wgt = __expf(sims[idx][tid] - mmax);
        sum += wgt;
        for (int c2 = 0; c2 < 32; ++c2) {
          v2h vv = kv[c2][ty + di + oy][tx + dj + ox];
          acc[2 * c2]     += wgt * (float)vv[0];
          acc[2 * c2 + 1] += wgt * (float)vv[1];
        }
      }
    float inv = 1.0f / sum;
    for (int c = 0; c < 64; ++c)
      out[(img * CCH + c) * HWSZ + p] = acc[c] * inv;
  }
}

// ---------------------------------------------------------------------------
extern "C" void kernel_launch(void* const* d_in, const int* in_sizes, int n_in,
                              void* d_out, int out_size, void* d_ws,
                              size_t ws_size, hipStream_t stream) {
  const float* img_feats = (const float*)d_in[0];
  const float* warped    = (const float*)d_in[1];
  const float* w_convs   = (const float*)d_in[2];
  const float* bn_g      = (const float*)d_in[3];
  const float* bn_b      = (const float*)d_in[4];
  const float* bn_m      = (const float*)d_in[5];
  const float* bn_v      = (const float*)d_in[6];
  (void)in_sizes; (void)n_in; (void)out_size; (void)ws_size;

  char* ws = (char*)d_ws;
  const size_t QKV_BYTES = (size_t)NIMG * 32 * HWSZ * sizeof(v2h);  // 17.2 MB
  half_t* wf   = (half_t*)(ws);
  float*  bias = (float*)(ws + 40960);
  v2h* qb = (v2h*)(ws + 65536);
  v2h* kb = (v2h*)(ws + 65536 + QKV_BYTES);
  v2h* vb = (v2h*)(ws + 65536 + 2 * QKV_BYTES);
  float* outp = (float*)d_out;

  fuse_weights_kernel<<<1, 256, 0, stream>>>(w_convs, bn_g, bn_b, bn_m, bn_v,
                                             wf, bias);
  conv_wmma_kernel<true><<<256, 256, 0, stream>>>(img_feats, wf, bias, qb, 0, 1);
  conv_wmma_kernel<true><<<256, 256, 0, stream>>>(warped, wf, bias, kb, 2, 3);
  conv_wmma_kernel<false><<<256, 256, 0, stream>>>(warped, wf, bias, vb, 4, -1);
  local_attn_kernel<<<dim3(25, 14, NIMG), 64, 0, stream>>>(qb, kb, vb, outp);
}
